// GraphTransformerLayer_69312182223535
// MI455X (gfx1250) — matrix operand order repaired and checked by reading, not compiled, not run on hardware
//
#include <hip/hip_runtime.h>
#include <hip/hip_bf16.h>
#include <math.h>

#define NN   50000
#define EE   800000
#define DD   128
#define HH   8
#define FF   512          // 4*D
#define MREP 5            // 3125 M-tiles = 5 * 625, exact

typedef __attribute__((ext_vector_type(16))) __bf16 v16bf;
typedef __attribute__((ext_vector_type(8)))  __bf16 v8bf;
typedef __attribute__((ext_vector_type(8)))  float  v8f;

// ---------------------------------------------------------------------------
// helpers
// ---------------------------------------------------------------------------
__device__ __forceinline__ float wave_sum32(float v) {
#pragma unroll
    for (int m = 16; m > 0; m >>= 1) v += __shfl_xor(v, m, 32);
    return v;
}

__device__ __forceinline__ void atomic_max_f32(float* addr, float v) {
    if (v >= 0.0f) atomicMax((int*)addr, __float_as_int(v));
    else           atomicMin((unsigned int*)addr, (unsigned int)__float_as_int(v));
}

// Load a 16-element bf16 fragment as two 16B vector loads.
// off2 = offset of the second 8-element half (A and B fragments: +8).
__device__ __forceinline__ v16bf load_frag(const __bf16* __restrict__ p, int off2) {
    v8bf lo = *(const v8bf*)(p);
    v8bf hi = *(const v8bf*)(p + off2);
    v16bf r;
#pragma unroll
    for (int i = 0; i < 8; ++i) { r[i] = lo[i]; r[8 + i] = hi[i]; }
    return r;
}

// ---------------------------------------------------------------------------
// init: smax = -inf, denom = 0, agg = 0   (ws is poisoned by the harness)
// ---------------------------------------------------------------------------
__global__ void init_kernel(float* __restrict__ smax, float* __restrict__ denom,
                            float* __restrict__ agg) {
    int i = blockIdx.x * blockDim.x + threadIdx.x;
    if (i < NN * HH) {
        smax[i]  = __int_as_float((int)0xff800000); // -inf
        denom[i] = 0.0f;
    }
    if (i < NN * DD) agg[i] = 0.0f;
}

// ---------------------------------------------------------------------------
// f32 -> bf16 conversion (activations, row-major preserved)
// ---------------------------------------------------------------------------
__global__ void cvt_bf16_kernel(const float* __restrict__ src, __bf16* __restrict__ dst, int n) {
    int i = blockIdx.x * blockDim.x + threadIdx.x;
    if (i < n) dst[i] = (__bf16)src[i];
}

// f32 [K x Nd] -> bf16 transposed [Nd x K]  (weights: makes B-fragments contiguous)
__global__ void cvt_bf16_T_kernel(const float* __restrict__ src, __bf16* __restrict__ dst,
                                  int K, int Nd) {
    int i = blockIdx.x * blockDim.x + threadIdx.x;
    if (i < K * Nd) {
        int k = i / Nd;
        int n = i - k * Nd;
        dst[(size_t)n * K + k] = (__bf16)src[i];
    }
}

// ---------------------------------------------------------------------------
// LayerNorm over D=128, one wave32 per row, output bf16 (A-operand for WMMA)
// ---------------------------------------------------------------------------
__global__ void ln_bf16_kernel(const float* __restrict__ x, const float* __restrict__ g,
                               const float* __restrict__ b, __bf16* __restrict__ out,
                               int nrows) {
    int row = blockIdx.x * blockDim.y + threadIdx.y;
    if (row >= nrows) return;
    int lane = threadIdx.x;
    int f = 4 * lane;
    const float4 xv = *(const float4*)(x + (size_t)row * DD + f);
    float s1 = xv.x + xv.y + xv.z + xv.w;
    float s2 = xv.x * xv.x + xv.y * xv.y + xv.z * xv.z + xv.w * xv.w;
    s1 = wave_sum32(s1);
    s2 = wave_sum32(s2);
    const float mean = s1 * (1.0f / DD);
    const float var  = s2 * (1.0f / DD) - mean * mean;
    const float rstd = rsqrtf(var + 1e-5f);
    __bf16* o = out + (size_t)row * DD + f;
    o[0] = (__bf16)((xv.x - mean) * rstd * g[f + 0] + b[f + 0]);
    o[1] = (__bf16)((xv.y - mean) * rstd * g[f + 1] + b[f + 1]);
    o[2] = (__bf16)((xv.z - mean) * rstd * g[f + 2] + b[f + 2]);
    o[3] = (__bf16)((xv.w - mean) * rstd * g[f + 3] + b[f + 3]);
}

// ---------------------------------------------------------------------------
// WMMA GEMM: C[M x Nd] = A[M x K](bf16, row-major) @ B (bf16, stored TRANSPOSED
// as Bt[Nd x K]) + bias.  One wave computes MREP M-tiles x one N-tile so each
// B fragment (2 x b128 loads) is reused across MREP WMMAs.
// act: 0 = none, 1 = exact GELU.  res (optional) is added AFTER bias/act.
// ---------------------------------------------------------------------------
__global__ void wmma_gemm_kernel(const __bf16* __restrict__ A, const __bf16* __restrict__ Bt,
                                 const float* __restrict__ bias,
                                 float* __restrict__ Cf, __bf16* __restrict__ Cb,
                                 const float* __restrict__ res,
                                 int M, int Ktot, int Nd, int act) {
    const int ntiles  = Nd >> 4;
    const int mgroups = (M >> 4) / MREP;          // 625
    const int total   = mgroups * ntiles;
    const int wid     = blockIdx.x * blockDim.y + threadIdx.y;
    if (wid >= total) return;
    const int mg   = wid / ntiles;
    const int nt   = wid % ntiles;
    const int lane = threadIdx.x;
    const int n    = lane & 15;
    const int hi   = lane >> 4;

    // Fragment layouts used here (contiguous k order, same for A and B; not the ISA table order):
    //  A 16x32: row = lane%16; lanes<16 -> K 0..15, lanes>=16 -> K 16..31
    //  B 32x16: col = lane%16; lanes<16 -> K 0..15, lanes>=16 -> K 16..31
    //  With Bt stored [Nd x K], each lane's B fragment is 16 contiguous bf16.
    const int aKbase = hi ? 16 : 0;
    const int bKbase = hi ? 16 : 0;

    const int mt0 = mg * MREP;
    size_t arow[MREP];
#pragma unroll
    for (int j = 0; j < MREP; ++j)
        arow[j] = (size_t)((mt0 + j) * 16 + n) * Ktot + aKbase;

    const __bf16* brow = Bt + (size_t)(nt * 16 + n) * Ktot + bKbase;

    const v8f zero = {};
    v8f acc[MREP];
#pragma unroll
    for (int j = 0; j < MREP; ++j) acc[j] = zero;

    for (int k0 = 0; k0 < Ktot; k0 += 32) {
        v16bf b = load_frag(brow + k0, 8);
#pragma unroll
        for (int j = 0; j < MREP; ++j) {
            v16bf a = load_frag(A + arow[j] + k0, 8);
            acc[j] = __builtin_amdgcn_wmma_f32_16x16x32_bf16(
                /*neg_a=*/false, a, /*neg_b=*/false, b,
                /*c_mod=*/(short)0, acc[j], /*reuse_a=*/false, /*reuse_b=*/false);
        }
    }

    const int col = nt * 16 + n;
    const float bvv = bias ? bias[col] : 0.0f;
#pragma unroll
    for (int j = 0; j < MREP; ++j) {
        const int mrow0 = (mt0 + j) * 16 + (hi ? 8 : 0);
#pragma unroll
        for (int r = 0; r < 8; ++r) {
            size_t idx = (size_t)(mrow0 + r) * Nd + col;
            float v = acc[j][r] + bvv;
            if (act == 1) v = 0.5f * v * (1.0f + erff(v * 0.70710678118f)); // exact GELU
            if (res) v += res[idx];
            if (Cf) Cf[idx] = v;
            if (Cb) Cb[idx] = (__bf16)v;
        }
    }
}

// ---------------------------------------------------------------------------
// Edge pass A: per-edge per-head scores + segment max.  One wave32 per edge.
// ---------------------------------------------------------------------------
__global__ void edge_scores_kernel(const long long* __restrict__ ei,
                                   const float* __restrict__ Q, const float* __restrict__ K,
                                   const float* __restrict__ ea, const float* __restrict__ We,
                                   const float* __restrict__ be,
                                   float* __restrict__ scores, float* __restrict__ smax) {
    int e = blockIdx.x * blockDim.y + threadIdx.y;
    if (e >= EE) return;
    const int lane = threadIdx.x;
    const long long r = ei[e];
    const long long c = ei[EE + e];
    const int f = 4 * lane;

    const float4 q = *(const float4*)(Q + (size_t)r * DD + f);
    const float4 k = *(const float4*)(K + (size_t)c * DD + f);
    const float4 a = *(const float4*)(ea + (size_t)e * DD + f);

    // per-head QK dot: reduce over the 4 lanes of each head group
    float p = q.x * k.x + q.y * k.y + q.z * k.z + q.w * k.w;
    p += __shfl_xor(p, 1, 32);
    p += __shfl_xor(p, 2, 32);

    // edge_attr @ We : each lane contributes its 4 features to all 8 heads
    float wh[HH];
    const float* w0 = We + (size_t)f * HH;
#pragma unroll
    for (int h = 0; h < HH; ++h)
        wh[h] = a.x * w0[h] + a.y * w0[HH + h] + a.z * w0[2 * HH + h] + a.w * w0[3 * HH + h];
#pragma unroll
    for (int m = 16; m > 0; m >>= 1) {
#pragma unroll
        for (int h = 0; h < HH; ++h) wh[h] += __shfl_xor(wh[h], m, 32);
    }

    float qk = __shfl(p, (lane & 7) * 4, 32);

    float wsel = wh[0];
#pragma unroll
    for (int h = 1; h < HH; ++h) if ((lane & 7) == h) wsel = wh[h];

    if (lane < HH) {
        float s = qk * 0.25f + wsel + be[lane];      // 1/sqrt(HD)=0.25
        scores[(size_t)e * HH + lane] = s;
        atomic_max_f32(smax + (size_t)r * HH + lane, s);
    }
}

// ---------------------------------------------------------------------------
// Edge pass B: ex = exp(score - smax[row]); accumulate denom. Thread per (e,h).
// ---------------------------------------------------------------------------
__global__ void edge_exp_kernel(const long long* __restrict__ ei,
                                float* __restrict__ scores,
                                const float* __restrict__ smax,
                                float* __restrict__ denom) {
    int i = blockIdx.x * blockDim.x + threadIdx.x;
    if (i >= EE * HH) return;
    int e = i >> 3;
    int h = i & 7;
    long long r = ei[e];
    float ex = __expf(scores[i] - smax[(size_t)r * HH + h]);
    scores[i] = ex;
    atomicAdd(denom + (size_t)r * HH + h, ex);
}

// ---------------------------------------------------------------------------
// Edge pass C: agg[row] += V[col] * (ex / denom[row]).  One wave32 per edge.
// ---------------------------------------------------------------------------
__global__ void edge_scatter_kernel(const long long* __restrict__ ei,
                                    const float* __restrict__ V,
                                    const float* __restrict__ scores,
                                    const float* __restrict__ denom,
                                    float* __restrict__ agg) {
    int e = blockIdx.x * blockDim.y + threadIdx.y;
    if (e >= EE) return;
    const int lane = threadIdx.x;
    const long long r = ei[e];
    const long long c = ei[EE + e];
    const int h = lane >> 2;
    const float alpha = scores[(size_t)e * HH + h] / denom[(size_t)r * HH + h];
    const int f = 4 * lane;
    const float4 v = *(const float4*)(V + (size_t)c * DD + f);
    float* dst = agg + (size_t)r * DD + f;
    atomicAdd(dst + 0, v.x * alpha);
    atomicAdd(dst + 1, v.y * alpha);
    atomicAdd(dst + 2, v.z * alpha);
    atomicAdd(dst + 3, v.w * alpha);
}

// ---------------------------------------------------------------------------
// launch
// ---------------------------------------------------------------------------
extern "C" void kernel_launch(void* const* d_in, const int* in_sizes, int n_in,
                              void* d_out, int out_size, void* d_ws, size_t ws_size,
                              hipStream_t stream) {
    const float*     x     = (const float*)d_in[0];
    const long long* ei    = (const long long*)d_in[1];
    const float*     ea    = (const float*)d_in[2];
    const float*     Wq    = (const float*)d_in[3];
    const float*     bq    = (const float*)d_in[4];
    const float*     Wk    = (const float*)d_in[5];
    const float*     bk    = (const float*)d_in[6];
    const float*     Wv    = (const float*)d_in[7];
    const float*     bv    = (const float*)d_in[8];
    const float*     Wo    = (const float*)d_in[9];
    const float*     bo    = (const float*)d_in[10];
    const float*     We    = (const float*)d_in[11];
    const float*     be    = (const float*)d_in[12];
    const float*     W1    = (const float*)d_in[13];
    const float*     bf1   = (const float*)d_in[14];
    const float*     W2    = (const float*)d_in[15];
    const float*     bf2   = (const float*)d_in[16];
    const float*     g1    = (const float*)d_in[17];
    const float*     beta1 = (const float*)d_in[18];
    const float*     g2    = (const float*)d_in[19];
    const float*     beta2 = (const float*)d_in[20];
    float* out = (float*)d_out;

    // ---- workspace carve-up (256B aligned slots) ----
    char* w = (char*)d_ws;
    size_t off = 0;
    auto take = [&](size_t bytes) -> char* {
        char* p = w + off;
        off += (bytes + 255) & ~(size_t)255;
        return p;
    };
    float* Qf     = (float*)take((size_t)NN * DD * 4);
    float* Kf     = (float*)take((size_t)NN * DD * 4);
    float* Vf     = (float*)take((size_t)NN * DD * 4);
    float* agg    = (float*)take((size_t)NN * DD * 4);
    float* x1     = (float*)take((size_t)NN * DD * 4);
    float* scores = (float*)take((size_t)EE * HH * 4);
    float* smax   = (float*)take((size_t)NN * HH * 4);
    float* denom  = (float*)take((size_t)NN * HH * 4);
    __bf16* xnb   = (__bf16*)take((size_t)NN * DD * 2);
    __bf16* aggb  = (__bf16*)take((size_t)NN * DD * 2);
    __bf16* xn2b  = (__bf16*)take((size_t)NN * DD * 2);
    __bf16* hb    = (__bf16*)take((size_t)NN * FF * 2);
    __bf16* wqb   = (__bf16*)take((size_t)DD * DD * 2);   // transposed [Nd x K]
    __bf16* wkb   = (__bf16*)take((size_t)DD * DD * 2);
    __bf16* wvb   = (__bf16*)take((size_t)DD * DD * 2);
    __bf16* wob   = (__bf16*)take((size_t)DD * DD * 2);
    __bf16* w1b   = (__bf16*)take((size_t)DD * FF * 2);   // [512 x 128]
    __bf16* w2b   = (__bf16*)take((size_t)FF * DD * 2);   // [128 x 512]

    const dim3 waveBlk(32, 8);   // 8 waves / block for per-row & per-edge kernels
    const dim3 gemmBlk(32, 4);   // 4 wave-tile-groups / block for WMMA GEMM

    // 0) init softmax tables + agg
    init_kernel<<<(NN * DD + 255) / 256, 256, 0, stream>>>(smax, denom, agg);

    // 1) weights -> bf16, transposed so B-fragments are K-contiguous per lane
    cvt_bf16_T_kernel<<<(DD * DD + 255) / 256, 256, 0, stream>>>(Wq, wqb, DD, DD);
    cvt_bf16_T_kernel<<<(DD * DD + 255) / 256, 256, 0, stream>>>(Wk, wkb, DD, DD);
    cvt_bf16_T_kernel<<<(DD * DD + 255) / 256, 256, 0, stream>>>(Wv, wvb, DD, DD);
    cvt_bf16_T_kernel<<<(DD * DD + 255) / 256, 256, 0, stream>>>(Wo, wob, DD, DD);
    cvt_bf16_T_kernel<<<(DD * FF + 255) / 256, 256, 0, stream>>>(W1, w1b, DD, FF);
    cvt_bf16_T_kernel<<<(FF * DD + 255) / 256, 256, 0, stream>>>(W2, w2b, FF, DD);

    // 2) LN1 -> bf16 activations
    ln_bf16_kernel<<<(NN + 7) / 8, waveBlk, 0, stream>>>(x, g1, beta1, xnb, NN);

    // 3) Q/K/V projections: 625 m-groups * 8 n-tiles = 5000 waves each
    const int qkvWaves = ((NN / 16) / MREP) * (DD / 16);
    const int qkvBlocks = (qkvWaves + 3) / 4;
    wmma_gemm_kernel<<<qkvBlocks, gemmBlk, 0, stream>>>(xnb, wqb, bq, Qf, nullptr, nullptr, NN, DD, DD, 0);
    wmma_gemm_kernel<<<qkvBlocks, gemmBlk, 0, stream>>>(xnb, wkb, bk, Kf, nullptr, nullptr, NN, DD, DD, 0);
    wmma_gemm_kernel<<<qkvBlocks, gemmBlk, 0, stream>>>(xnb, wvb, bv, Vf, nullptr, nullptr, NN, DD, DD, 0);

    // 4) edge scores + segment max   (Q/K rows hit the 192MB L2)
    edge_scores_kernel<<<(EE + 7) / 8, waveBlk, 0, stream>>>(ei, Qf, Kf, ea, We, be, scores, smax);

    // 5) exp + segment sum
    edge_exp_kernel<<<(EE * HH + 255) / 256, 256, 0, stream>>>(ei, scores, smax, denom);

    // 6) weighted scatter-add aggregation
    edge_scatter_kernel<<<(EE + 7) / 8, waveBlk, 0, stream>>>(ei, Vf, scores, denom, agg);

    // 7) output projection + residual:  x1 = x + agg @ Wo + bo
    cvt_bf16_kernel<<<(NN * DD + 255) / 256, 256, 0, stream>>>(agg, aggb, NN * DD);
    wmma_gemm_kernel<<<qkvBlocks, gemmBlk, 0, stream>>>(aggb, wob, bo, x1, nullptr, x, NN, DD, DD, 0);

    // 8) LN2 -> bf16
    ln_bf16_kernel<<<(NN + 7) / 8, waveBlk, 0, stream>>>(x1, g2, beta2, xn2b, NN);

    // 9) FFN: h = gelu(xn2 @ W1 + bf1)  (bf16 out, feeds next GEMM)
    const int ffn1Waves = ((NN / 16) / MREP) * (FF / 16);
    const int ffn1Blocks = (ffn1Waves + 3) / 4;
    wmma_gemm_kernel<<<ffn1Blocks, gemmBlk, 0, stream>>>(xn2b, w1b, bf1, nullptr, hb, nullptr, NN, DD, FF, 1);

    // 10) out = x1 + h @ W2 + bf2
    wmma_gemm_kernel<<<qkvBlocks, gemmBlk, 0, stream>>>(hb, w2b, bf2, out, nullptr, x1, NN, FF, DD, 0);
}
